// GatedDeltaNet_23003844837999
// MI455X (gfx1250) — matrix operand order repaired and checked by reading, not compiled, not run on hardware
//
#include <hip/hip_runtime.h>
#include <hip/hip_bf16.h>

// Problem constants (match reference)
#define B_      2
#define T_      2048
#define DM_     2048
#define H_      16
#define D_      128
#define M_      (B_ * T_)      // 4096 rows of x
#define K_      DM_            // 2048 reduction dim
#define INNER_  (H_ * D_)      // 2048
#define SROW_   132            // padded LDS row pitch (floats) -> no bank conflicts
#define STG_    (4 * D_ + 8)   // staging buffer stride in floats (k,q,v,a,beta)

typedef __bf16 bf16_t;
typedef __attribute__((ext_vector_type(16))) __bf16 v16bf;
typedef __attribute__((ext_vector_type(8)))  __bf16 v8bf;
typedef __attribute__((ext_vector_type(8)))  float  v8f;

// CDNA5 async global->LDS path (ASYNCcnt), if the toolchain exposes it.
#if defined(__has_builtin)
#  if __has_builtin(__builtin_amdgcn_global_load_async_to_lds_b32) && \
      __has_builtin(__builtin_amdgcn_s_wait_asynccnt)
#    define GDN_ASYNC 1
#  endif
#endif

// Generic -> AS1/AS3 pointer construction (builtin params are int* in the
// respective address spaces; clang printed AS1 as '__device__ int *').
// Flat LDS addresses are {SHARED_BASE, lds_offset}; truncating to 32 bits
// yields the LDS offset (ISA 10.2 aperture rules), so inttoptr is safe.
#define GDN_AS1(p) ((__attribute__((address_space(1))) int*)(uintptr_t)(p))
#define GDN_AS3(p) ((__attribute__((address_space(3))) int*)(unsigned)(uintptr_t)(p))

// ---------------------------------------------------------------------------
// f32 -> bf16 conversion
// ---------------------------------------------------------------------------
__global__ void gdn_cvt_bf16(const float* __restrict__ src,
                             bf16_t* __restrict__ dst, int n) {
  int i = blockIdx.x * blockDim.x + threadIdx.x;
  if (i < n) dst[i] = (bf16_t)src[i];
}

// ---------------------------------------------------------------------------
// C(M,N) = A(M,K) @ W(N,K)^T, bf16 in / f32 accum, v_wmma_f32_16x16x32_bf16.
// Register-blocked: each wave owns a (16*MT) x (16*NT) tile -> MT*NT
// accumulators, MT A-fragments and NT B-fragments per 32-wide K step
// (8 independent WMMAs per step for MT=2,NT=4; ~21 flops/byte from L2).
// Epilogue: optional bias[col] + sigmoid.
// ---------------------------------------------------------------------------
template <int MT, int NT>
__global__ __launch_bounds__(256)
void gdn_gemm_bf16_wmma(const bf16_t* __restrict__ A,
                        const bf16_t* __restrict__ W,
                        float* __restrict__ C,
                        int Mq, int Nq, int Kq,
                        const float* __restrict__ bias, int act)
{
  const int lane = threadIdx.x & 31;
  const int wid  = threadIdx.x >> 5;
  const int tiles_n = Nq / (16 * NT);
  const int tiles   = (Mq / (16 * MT)) * tiles_n;
  const int gw = blockIdx.x * 8 + wid;
  if (gw >= tiles) return;                 // wave-uniform exit
  const int tm   = (gw / tiles_n) * (16 * MT);
  const int tn   = (gw % tiles_n) * (16 * NT);
  const int half = lane >> 4;
  const int l15  = lane & 15;

  // A fragment: row = tm+i*16+l15; elems 0..7 -> K=half*8+{0..7}, 8..15 -> +16
  const bf16_t* arow[MT];
  // B fragment (= W^T): col = tn+j*16+l15; elems 0..7 -> K=half*8+{0..7}, 8..15 -> +16
  const bf16_t* brow[NT];
#pragma unroll
  for (int i = 0; i < MT; ++i)
    arow[i] = A + (size_t)(tm + i * 16 + l15) * Kq + half * 8;
#pragma unroll
  for (int j = 0; j < NT; ++j)
    brow[j] = W + (size_t)(tn + j * 16 + l15) * Kq + half * 8;

  v8f acc[MT][NT] = {};
  for (int kk = 0; kk < Kq; kk += 32) {
    v16bf afr[MT], bfr[NT];
#pragma unroll
    for (int i = 0; i < MT; ++i) {
      v8bf lo = *(const v8bf*)(arow[i] + kk);
      v8bf hi = *(const v8bf*)(arow[i] + kk + 16);
      afr[i] = __builtin_shufflevector(lo, hi, 0,1,2,3,4,5,6,7,
                                       8,9,10,11,12,13,14,15);
      __builtin_prefetch(arow[i] + kk + 512, 0, 3);
    }
#pragma unroll
    for (int j = 0; j < NT; ++j) {
      v8bf lo = *(const v8bf*)(brow[j] + kk);
      v8bf hi = *(const v8bf*)(brow[j] + kk + 16);
      bfr[j] = __builtin_shufflevector(lo, hi, 0,1,2,3,4,5,6,7,
                                       8,9,10,11,12,13,14,15);
      __builtin_prefetch(brow[j] + kk + 512, 0, 3);
    }
#pragma unroll
    for (int i = 0; i < MT; ++i)
#pragma unroll
      for (int j = 0; j < NT; ++j)
        acc[i][j] = __builtin_amdgcn_wmma_f32_16x16x32_bf16(
            false, afr[i], false, bfr[j], (short)0, acc[i][j], false, false);
  }

#pragma unroll
  for (int j = 0; j < NT; ++j) {
    const int col = tn + j * 16 + l15;
    const float bv = bias ? bias[col] : 0.0f;
#pragma unroll
    for (int i = 0; i < MT; ++i) {
#pragma unroll
      for (int r = 0; r < 8; ++r) {
        int row = tm + i * 16 + r + 8 * half;   // C layout: halves split M
        float val = acc[i][j][r] + bv;
        if (act) val = 1.0f / (1.0f + __expf(-val));
        C[(size_t)row * Nq + col] = val;
      }
    }
  }
}

// ---------------------------------------------------------------------------
// L2-normalize rows of length 128 in place. One wave per row (wave32).
// ---------------------------------------------------------------------------
__global__ __launch_bounds__(256)
void gdn_l2norm_rows(float* __restrict__ buf, int nrows) {
  const int lane = threadIdx.x & 31;
  const int wid  = threadIdx.x >> 5;
  const int row  = blockIdx.x * 8 + wid;
  if (row >= nrows) return;
  float* p = buf + (size_t)row * D_;
  float4 v = *(float4*)(p + lane * 4);
  float ss = v.x * v.x + v.y * v.y + v.z * v.z + v.w * v.w;
  ss += __shfl_xor(ss, 16);
  ss += __shfl_xor(ss, 8);
  ss += __shfl_xor(ss, 4);
  ss += __shfl_xor(ss, 2);
  ss += __shfl_xor(ss, 1);
  float s = 1.0f / fmaxf(sqrtf(ss), 1e-12f);
  v.x *= s; v.y *= s; v.z *= s; v.w *= s;
  *(float4*)(p + lane * 4) = v;
}

// ---------------------------------------------------------------------------
// Sequential delta-rule recurrence. One block per (b,h); 128 threads; the
// 128x128 f32 state lives in (dynamic) LDS. Thread i owns state row i.
// The next step's k/q/v/alpha/beta vectors are double-buffered into LDS with
// GLOBAL_LOAD_ASYNC_TO_LDS_B32 (ASYNCcnt) so the global fetch overlaps the
// two state sweeps of the current step.
// ---------------------------------------------------------------------------
__global__ __launch_bounds__(128)
void gdn_recurrence(const float* __restrict__ q,
                    const float* __restrict__ k,
                    const float* __restrict__ v,
                    const float* __restrict__ a,
                    const float* __restrict__ beta,
                    const float* __restrict__ state0,
                    float* __restrict__ outp)
{
  extern __shared__ float sm[];
  float* S     = sm;                  // D_ * SROW_
  float* stage = sm + D_ * SROW_;     // 2 * STG_ (k,q,v,a vectors + beta)

  const int bh = blockIdx.x;          // b*H + h
  const int b  = bh / H_;
  const int h  = bh % H_;
  const int i  = threadIdx.x;         // state row

  // init S from input state (B,H,D,D)
  const float* s0 = state0 + ((size_t)bh * D_ + i) * D_;
  float* Srow = S + i * SROW_;
  for (int j = 0; j < D_; j += 4)
    *(float4*)(Srow + j) = *(const float4*)(s0 + j);

  auto stage_step = [&](int t, float* buf) {
    const size_t idx  = ((size_t)(b * T_ + t) * H_ + h) * D_ + i;
    const size_t bidx = (size_t)(b * T_ + t) * H_ + h;
#ifdef GDN_ASYNC
    __builtin_amdgcn_global_load_async_to_lds_b32(GDN_AS1(k + idx),
                                                  GDN_AS3(buf + i), 0, 0);
    __builtin_amdgcn_global_load_async_to_lds_b32(GDN_AS1(q + idx),
                                                  GDN_AS3(buf + D_ + i), 0, 0);
    __builtin_amdgcn_global_load_async_to_lds_b32(GDN_AS1(v + idx),
                                                  GDN_AS3(buf + 2 * D_ + i), 0, 0);
    __builtin_amdgcn_global_load_async_to_lds_b32(GDN_AS1(a + idx),
                                                  GDN_AS3(buf + 3 * D_ + i), 0, 0);
    if (i == 0)
      __builtin_amdgcn_global_load_async_to_lds_b32(GDN_AS1(beta + bidx),
                                                    GDN_AS3(buf + 4 * D_), 0, 0);
#else
    buf[i]          = k[idx];
    buf[D_ + i]     = q[idx];
    buf[2 * D_ + i] = v[idx];
    buf[3 * D_ + i] = a[idx];
    if (i == 0) buf[4 * D_] = beta[bidx];
#endif
  };
  auto wait_stage = [&]() {
#ifdef GDN_ASYNC
    __builtin_amdgcn_s_wait_asynccnt(0);
#endif
    __syncthreads();
  };

  int cur = 0;
  stage_step(0, stage);
  wait_stage();

  for (int t = 0; t < T_; ++t) {
    float* buf = stage + cur * STG_;
    if (t + 1 < T_) stage_step(t + 1, stage + (cur ^ 1) * STG_);

    const float* kv = buf;
    const float* qv = buf + D_;
    const float* vv = buf + 2 * D_;
    const float* av = buf + 3 * D_;
    const float  bt = buf[4 * D_];

    // Sk_i = dot(S[i,:], k)
    float sk = 0.f;
    for (int j = 0; j < D_; j += 4) {
      float4 s4 = *(float4*)(Srow + j);
      float4 k4 = *(const float4*)(kv + j);
      sk += s4.x * k4.x + s4.y * k4.y + s4.z * k4.z + s4.w * k4.w;
    }
    const float coef = bt * (vv[i] - sk);   // beta * (v - Sk)
    const float ai   = av[i];

    // S[i,:] = ai*S[i,:] + coef*k ;  out_i = dot(S_new[i,:], q)
    float o = 0.f;
    for (int j = 0; j < D_; j += 4) {
      float4 s4 = *(float4*)(Srow + j);
      float4 k4 = *(const float4*)(kv + j);
      float4 q4 = *(const float4*)(qv + j);
      s4.x = fmaf(ai, s4.x, coef * k4.x);
      s4.y = fmaf(ai, s4.y, coef * k4.y);
      s4.z = fmaf(ai, s4.z, coef * k4.z);
      s4.w = fmaf(ai, s4.w, coef * k4.w);
      *(float4*)(Srow + j) = s4;
      o += s4.x * q4.x + s4.y * q4.y + s4.z * q4.z + s4.w * q4.w;
    }
    outp[((size_t)(b * T_ + t) * H_ + h) * D_ + i] = o;

    wait_stage();   // next buffer resident; all reads of current buffer done
    cur ^= 1;
  }
}

// ---------------------------------------------------------------------------
// y = bf16(out_pre * g)
// ---------------------------------------------------------------------------
__global__ void gdn_gate_cvt(const float* __restrict__ o,
                             const float* __restrict__ g,
                             bf16_t* __restrict__ y, int n) {
  int i = blockIdx.x * blockDim.x + threadIdx.x;
  if (i < n) y[i] = (bf16_t)(o[i] * g[i]);
}

// ---------------------------------------------------------------------------
extern "C" void kernel_launch(void* const* d_in, const int* in_sizes, int n_in,
                              void* d_out, int out_size, void* d_ws, size_t ws_size,
                              hipStream_t stream)
{
  (void)in_sizes; (void)n_in; (void)out_size; (void)ws_size;
  const float* x  = (const float*)d_in[0];
  const float* st = (const float*)d_in[1];
  const float* Wq = (const float*)d_in[2];
  const float* Wk = (const float*)d_in[3];
  const float* Wv = (const float*)d_in[4];
  const float* Wa = (const float*)d_in[5];
  const float* ba = (const float*)d_in[6];
  const float* Wb = (const float*)d_in[7];
  const float* bb = (const float*)d_in[8];
  const float* Wg = (const float*)d_in[9];
  const float* bg = (const float*)d_in[10];
  const float* Wo = (const float*)d_in[11];
  float* out = (float*)d_out;

  char* ws = (char*)d_ws;
  size_t off = 0;
  auto wsalloc = [&](size_t bytes) -> void* {
    void* p = ws + off;
    off = (off + bytes + 255) & ~(size_t)255;
    return p;
  };

  bf16_t* xb   = (bf16_t*)wsalloc((size_t)M_ * K_ * 2);
  bf16_t* wqb  = (bf16_t*)wsalloc((size_t)INNER_ * K_ * 2);
  bf16_t* wkb  = (bf16_t*)wsalloc((size_t)INNER_ * K_ * 2);
  bf16_t* wvb  = (bf16_t*)wsalloc((size_t)INNER_ * K_ * 2);
  bf16_t* wab  = (bf16_t*)wsalloc((size_t)INNER_ * K_ * 2);
  bf16_t* wgb  = (bf16_t*)wsalloc((size_t)INNER_ * K_ * 2);
  bf16_t* wob  = (bf16_t*)wsalloc((size_t)DM_ * INNER_ * 2);
  bf16_t* wbb  = (bf16_t*)wsalloc((size_t)H_ * K_ * 2);
  float* qf    = (float*)wsalloc((size_t)M_ * INNER_ * 4);
  float* kf    = (float*)wsalloc((size_t)M_ * INNER_ * 4);
  float* vf    = (float*)wsalloc((size_t)M_ * INNER_ * 4);
  float* af    = (float*)wsalloc((size_t)M_ * INNER_ * 4);
  float* gf    = (float*)wsalloc((size_t)M_ * INNER_ * 4);
  float* betaf = (float*)wsalloc((size_t)M_ * H_ * 4);
  float* op    = (float*)wsalloc((size_t)M_ * INNER_ * 4);
  bf16_t* yb   = xb;   // x no longer needed after projections; reuse buffer

  auto cvt = [&](const float* s, bf16_t* d, int n) {
    gdn_cvt_bf16<<<dim3((n + 255) / 256), dim3(256), 0, stream>>>(s, d, n);
  };
  cvt(x,  xb,  M_ * K_);
  cvt(Wq, wqb, INNER_ * K_);
  cvt(Wk, wkb, INNER_ * K_);
  cvt(Wv, wvb, INNER_ * K_);
  cvt(Wa, wab, INNER_ * K_);
  cvt(Wg, wgb, INNER_ * K_);
  cvt(Wo, wob, DM_ * INNER_);
  cvt(Wb, wbb, H_ * K_);

  // Big GEMMs: 32x64 per wave (2x4 WMMA tiles).
  auto gemm24 = [&](const bf16_t* A, const bf16_t* W, float* C, int Mq, int Nq,
                    const float* bias, int act) {
    int tiles = (Mq / 32) * (Nq / 64);
    gdn_gemm_bf16_wmma<2, 4><<<dim3((tiles + 7) / 8), dim3(256), 0, stream>>>(
        A, W, C, Mq, Nq, K_, bias, act);
  };
  // Narrow GEMM (beta, N=16): 16x16 per wave.
  auto gemm11 = [&](const bf16_t* A, const bf16_t* W, float* C, int Mq, int Nq,
                    const float* bias, int act) {
    int tiles = (Mq / 16) * (Nq / 16);
    gdn_gemm_bf16_wmma<1, 1><<<dim3((tiles + 7) / 8), dim3(256), 0, stream>>>(
        A, W, C, Mq, Nq, K_, bias, act);
  };

  gemm24(xb, wqb, qf,    M_, INNER_, nullptr, 0);
  gemm24(xb, wkb, kf,    M_, INNER_, nullptr, 0);
  gemm24(xb, wvb, vf,    M_, INNER_, nullptr, 0);
  gemm24(xb, wab, af,    M_, INNER_, ba, 1);        // sigmoid(x Wa^T + ba)
  gemm11(xb, wbb, betaf, M_, H_,     bb, 1);        // sigmoid(x Wb^T + bb)
  gemm24(xb, wgb, gf,    M_, INNER_, bg, 1);        // sigmoid(x Wg^T + bg)

  const int nrows = M_ * H_;   // (m,h) rows of length D are contiguous
  gdn_l2norm_rows<<<dim3((nrows + 7) / 8), dim3(256), 0, stream>>>(qf, nrows);
  gdn_l2norm_rows<<<dim3((nrows + 7) / 8), dim3(256), 0, stream>>>(kf, nrows);

  const size_t smem = (size_t)(D_ * SROW_ + 2 * STG_) * sizeof(float);
  gdn_recurrence<<<dim3(B_ * H_), dim3(D_), smem, stream>>>(
      qf, kf, vf, af, betaf, st, op);

  gdn_gate_cvt<<<dim3((M_ * INNER_ + 255) / 256), dim3(256), 0, stream>>>(
      op, gf, yb, M_ * INNER_);

  gemm24(yb, wob, out, M_, DM_, nullptr, 0);
}
